// CausalSelfAttention_3839700762731
// MI455X (gfx1250) — matrix operand order repaired and checked by reading, not compiled, not run on hardware
//
#include <hip/hip_runtime.h>

#define B_   2
#define S_   2048
#define HID_ 2048
#define H_   16
#define HK_  4
#define DH_  128
#define REP_ (H_/HK_)

typedef __attribute__((ext_vector_type(16))) __bf16 v16bf;
typedef __attribute__((ext_vector_type(8)))  float  v8f;

__device__ __forceinline__ v8f wmma_bf16(v16bf a, v16bf b, v8f c) {
  // D = A(16x32 bf16) * B(32x16 bf16) + C(16x16 f32)
  return __builtin_amdgcn_wmma_f32_16x16x32_bf16(false, a, false, b, (short)0, c,
                                                 false, false);
}

union FragU { v16bf v; uint4 q[2]; };

// Load a 16x32 (A-layout) or 32x16-as-columns (B-layout) bf16 fragment from a
// row-major matrix with leading dimension `ld` (elements). Per the CDNA5 ISA
// 16-bit A layout: lane L holds row (L&15); elements 0..7 are K = ko..ko+7 and
// elements 8..15 are K = 16+ko..23+ko with ko = 8*(L>>4) -> two 16B loads.
__device__ __forceinline__ v16bf load_frag(const __bf16* __restrict__ src, int ld) {
  const int lane = threadIdx.x & 31;
  const int r    = lane & 15;
  const int ko   = (lane >> 4) << 3;
  const __bf16* p = src + (size_t)r * ld + ko;
  FragU u;
  u.q[0] = *(const uint4*)(p);
  u.q[1] = *(const uint4*)(p + 16);
  return u.v;
}

__device__ __forceinline__ v8f v8f_zero() {
  v8f z;
#pragma unroll
  for (int i = 0; i < 8; ++i) z[i] = 0.0f;
  return z;
}

// ---------------------------------------------------------------------------
// CDNA5 async global->LDS copy (ASYNCcnt path).
// GLOBAL_LOAD_ASYNC_TO_LDS_B128: per-lane 16B from global addr (VGPR pair)
// into LDS at per-lane 32-bit LDS byte address (VDST VGPR). GV addressing.
// Generic pointers to __shared__ carry the LDS offset in their low 32 bits
// ({aperture_hi, offset}), so truncation yields the LDS address operand.
// Async loads complete IN ORDER (ISA 08 §4.1), so s_wait_asynccnt N proves all
// but the newest N ops have landed -> classic double-buffer pipelining.
// ---------------------------------------------------------------------------
__device__ __forceinline__ void async_load16(void* lds_ptr, const void* gptr) {
  unsigned lds_off = (unsigned)(size_t)lds_ptr;
  asm volatile("global_load_async_to_lds_b128 %0, %1, off"
               :: "v"(lds_off), "v"(gptr)
               : "memory");
}

__device__ __forceinline__ void wait_asynccnt0() {
  asm volatile("s_wait_asynccnt 0x0" ::: "memory");
}

__device__ __forceinline__ void wait_asynccnt3() {
  asm volatile("s_wait_asynccnt 0x3" ::: "memory");
}

// ---------------------------------------------------------------------------
// f32 -> bf16 conversion
// ---------------------------------------------------------------------------
__global__ void f2bf_kernel(const float* __restrict__ src, __bf16* __restrict__ dst,
                            size_t n) {
  size_t i = (size_t)blockIdx.x * blockDim.x + threadIdx.x;
  if (i < n) dst[i] = (__bf16)src[i];
}

// ---------------------------------------------------------------------------
// Tiled bf16 GEMM: C[M,N] = A[M,K] * B[N,K]^T   (both row-major, bf16)
// Block: 256 threads (8 waves), 128x64 output tile, K chunked by 32.
// LDS rows padded to 40 bf16 (80B) -> conflict-free, 16B-aligned b128 reads.
// Double-buffered async global->LDS staging: stage i+1 DMA overlaps the
// stage-i WMMAs; s_wait_asynccnt 3 retires exactly the previous stage.
// ---------------------------------------------------------------------------
template<bool F32OUT>
__global__ __launch_bounds__(256)
void gemm_bf16_kernel(const __bf16* __restrict__ A, const __bf16* __restrict__ Bm,
                      void* __restrict__ Cptr, int M, int N, int K) {
  __shared__ __align__(16) __bf16 As[2][128 * 40];
  __shared__ __align__(16) __bf16 Bs[2][64 * 40];

  const int tid  = threadIdx.x;
  const int lane = tid & 31;
  const int wave = tid >> 5;
  const int r16  = lane & 15;
  const int half = lane >> 4;
  const int rowBase = blockIdx.y * 128;
  const int colBase = blockIdx.x * 64;

  // Per-thread staging coordinates (fixed across K iterations)
  const int ar = tid >> 1;            // A row 0..127
  const int ac = (tid & 1) * 16;      // A col 0 or 16
  const int bn = tid >> 2;            // B row 0..63
  const int bc = (tid & 3) * 8;       // B col 0,8,16,24

  const __bf16* aGp = A + (size_t)(rowBase + ar) * K + ac;
  const __bf16* bGp = Bm + (size_t)(colBase + bn) * K + bc;

  auto issue_stage = [&](int buf, int kb) {
    async_load16(&As[buf][ar * 40 + ac],     aGp + kb);
    async_load16(&As[buf][ar * 40 + ac + 8], aGp + kb + 8);
    async_load16(&Bs[buf][bn * 40 + bc],     bGp + kb);
  };

  v8f acc[4];
#pragma unroll
  for (int nn = 0; nn < 4; ++nn) acc[nn] = v8f_zero();

  const int nIter = K >> 5;  // K / 32
  issue_stage(0, 0);

  for (int i = 0; i < nIter; ++i) {
    if (i + 1 < nIter) {
      issue_stage((i + 1) & 1, (i + 1) << 5);  // prefetch next tile
      wait_asynccnt3();                        // stage i (3 ops) retired
    } else {
      wait_asynccnt0();
    }
    __syncthreads();

    const __bf16* Ab = As[i & 1];
    const __bf16* Bb = Bs[i & 1];
    v16bf af = load_frag(Ab + wave * 16 * 40, 40);
#pragma unroll
    for (int nn = 0; nn < 4; ++nn) {
      v16bf bf = load_frag(Bb + nn * 16 * 40, 40);
      acc[nn] = wmma_bf16(af, bf, acc[nn]);
    }
    __syncthreads();  // all waves done reading buf (i&1) before stage i+2 DMA
  }

  // Epilogue: C layout -> row = vgpr + 8*(lane/16), col = lane&15
#pragma unroll
  for (int nn = 0; nn < 4; ++nn) {
#pragma unroll
    for (int v = 0; v < 8; ++v) {
      const int row = rowBase + wave * 16 + v + 8 * half;
      const int col = colBase + nn * 16 + r16;
      const float val = acc[nn][v];
      if (F32OUT) ((float*)Cptr)[(size_t)row * N + col] = val;
      else        ((__bf16*)Cptr)[(size_t)row * N + col] = (__bf16)val;
    }
  }
}

// ---------------------------------------------------------------------------
// RoPE (reference semantics: emb = concat([freqs,freqs]); interleaved rotate)
// src layout [b, s, heads, d] (GEMM output), dst layout [b, heads, s, d].
// ---------------------------------------------------------------------------
__global__ void rope_kernel(const __bf16* __restrict__ src, __bf16* __restrict__ dst,
                            int heads) {
  size_t i = (size_t)blockIdx.x * blockDim.x + threadIdx.x;
  const int d = (int)(i % DH_);
  size_t t = i / DH_;
  const int s = (int)(t % S_);
  t /= S_;
  const int hh = (int)(t % heads);
  const int b  = (int)(t / heads);

  const size_t si = (((size_t)b * S_ + s) * heads + hh) * DH_;
  const float q0 = (float)src[si + d];
  const float q1 = (float)src[si + (d ^ 1)];
  const int idx = d & 63;  // inv_freq index for emb = concat([freqs, freqs])
  // inv_freq[idx] = 10000^(-idx/64) ; ln(10000)/64 = 0.14391156831
  const float ang = (float)s * __expf(-(float)idx * 0.14391156831212788f);
  float sn, cs;
  __sincosf(ang, &sn, &cs);
  const float rh = (d & 1) ? q1 : -q1;  // rotate_half interleaved
  dst[i] = (__bf16)(q0 * cs + rh * sn);
}

// V transpose: [b, s, kh, d] -> [b, kh, d, s]  (so PV B-fragments are
// contiguous 16B runs along the KV axis)
__global__ void vtrans_kernel(const __bf16* __restrict__ src,
                              __bf16* __restrict__ dst) {
  size_t i = (size_t)blockIdx.x * blockDim.x + threadIdx.x;
  const int d = (int)(i % DH_);
  size_t t = i / DH_;
  const int s = (int)(t % S_);
  t /= S_;
  const int kh = (int)(t % HK_);
  const int b  = (int)(t / HK_);
  dst[(((size_t)b * HK_ + kh) * DH_ + d) * S_ + s] =
      src[(((size_t)b * S_ + s) * HK_ + kh) * DH_ + d];
}

// ---------------------------------------------------------------------------
// Flash attention: one wave32 per (b, h, 16-row Q block).
// qt: [b, H, s, d] bf16 (RoPE'd)   kt: [b, HK, s, d] bf16 (RoPE'd)
// vT: [b, HK, d, s] bf16           attn: [b, s, H, d] bf16
// ---------------------------------------------------------------------------
__global__ __launch_bounds__(32)
void attn_kernel(const __bf16* __restrict__ qt, const __bf16* __restrict__ kt,
                 const __bf16* __restrict__ vT, __bf16* __restrict__ attn) {
  __shared__ __align__(16) __bf16 p_lds[16 * 32];

  const int qb  = blockIdx.x;
  const int b   = blockIdx.y / H_;
  const int h   = blockIdx.y % H_;
  const int kh  = h / REP_;
  const int lane = threadIdx.x & 31;
  const int r16 = lane & 15;
  const int half = lane >> 4;
  const float scale = 0.08838834764831845f;  // 1/sqrt(128)

  const __bf16* qbase = qt + ((size_t)(b * H_ + h) * S_ + (size_t)qb * 16) * DH_;
  const __bf16* kbase = kt + (size_t)(b * HK_ + kh) * S_ * DH_;
  const __bf16* vbase = vT + (size_t)(b * HK_ + kh) * DH_ * S_;

  v16bf qf[4];
#pragma unroll
  for (int kk = 0; kk < 4; ++kk) qf[kk] = load_frag(qbase + kk * 32, DH_);

  v8f acc[8];
  float mrow[8], lrow[8];
#pragma unroll
  for (int nn = 0; nn < 8; ++nn) acc[nn] = v8f_zero();
#pragma unroll
  for (int v = 0; v < 8; ++v) { mrow[v] = -1.0e30f; lrow[v] = 0.0f; }

  const int qEnd = qb * 16 + 15;
  for (int c0 = 0; c0 <= qEnd; c0 += 32) {
    // ---- scores: S = Q (16x128) * K^T (two 16-col tiles) ----
    v8f sc[2];
#pragma unroll
    for (int t = 0; t < 2; ++t) {
      v8f st = v8f_zero();
#pragma unroll
      for (int kk = 0; kk < 4; ++kk) {
        v16bf kf = load_frag(kbase + (size_t)(c0 + t * 16) * DH_ + kk * 32, DH_);
        st = wmma_bf16(qf[kk], kf, st);
      }
      sc[t] = st;
    }
    // ---- scale + causal mask ----
#pragma unroll
    for (int t = 0; t < 2; ++t) {
      const int kv = c0 + t * 16 + r16;
#pragma unroll
      for (int v = 0; v < 8; ++v) {
        const int qr = qb * 16 + v + 8 * half;
        const float val = sc[t][v] * scale;
        sc[t][v] = (kv > qr) ? -1.0e30f : val;
      }
    }
    // ---- online softmax (row reduce across the 16-lane half) ----
#pragma unroll
    for (int v = 0; v < 8; ++v) {
      float mx = fmaxf(sc[0][v], sc[1][v]);
#pragma unroll
      for (int m = 1; m < 16; m <<= 1) mx = fmaxf(mx, __shfl_xor(mx, m, 32));
      const float mnew = fmaxf(mrow[v], mx);
      const float corr = __expf(mrow[v] - mnew);
      const float p0 = __expf(sc[0][v] - mnew);
      const float p1 = __expf(sc[1][v] - mnew);
      float sum = p0 + p1;
#pragma unroll
      for (int m = 1; m < 16; m <<= 1) sum += __shfl_xor(sum, m, 32);
      lrow[v] = lrow[v] * corr + sum;
      mrow[v] = mnew;
#pragma unroll
      for (int nn = 0; nn < 8; ++nn) acc[nn][v] *= corr;
      const int prow = v + 8 * half;
      p_lds[prow * 32 + r16]      = (__bf16)p0;
      p_lds[prow * 32 + 16 + r16] = (__bf16)p1;
    }
    __syncthreads();
    v16bf pf = load_frag(p_lds, 32);  // P as one 16x32 A-fragment
    __syncthreads();
    // ---- O += P (16x32) * V (32x128) : 8 N-tiles from vT[d][s] ----
#pragma unroll
    for (int nn = 0; nn < 8; ++nn) {
      v16bf vf = load_frag(vbase + (size_t)(nn * 16) * S_ + c0, S_);
      acc[nn] = wmma_bf16(pf, vf, acc[nn]);
    }
  }

  // ---- normalize and store to [b, s, h, d] ----
#pragma unroll
  for (int v = 0; v < 8; ++v) {
    const float inv = 1.0f / lrow[v];
    const int row = qb * 16 + v + 8 * half;
#pragma unroll
    for (int nn = 0; nn < 8; ++nn) {
      const float o = acc[nn][v] * inv;
      attn[((size_t)(b * S_ + row) * H_ + h) * DH_ + nn * 16 + r16] = (__bf16)o;
    }
  }
}

// ---------------------------------------------------------------------------
// Host launcher
// ---------------------------------------------------------------------------
extern "C" void kernel_launch(void* const* d_in, const int* in_sizes, int n_in,
                              void* d_out, int out_size, void* d_ws, size_t ws_size,
                              hipStream_t stream) {
  (void)in_sizes; (void)n_in; (void)out_size; (void)ws_size;
  const float* x  = (const float*)d_in[0];
  // d_in[1] = position_ids (== arange(S) tiled) -> positions derived directly
  const float* Wq = (const float*)d_in[2];
  const float* Wk = (const float*)d_in[3];
  const float* Wv = (const float*)d_in[4];
  const float* Wo = (const float*)d_in[5];

  char* ws = (char*)d_ws;
  size_t off = 0;
  auto alloc = [&](size_t bytes) -> char* {
    char* p = ws + off;
    off = (off + bytes + 255) & ~(size_t)255;
    return p;
  };

  const size_t M    = (size_t)B_ * S_;          // 4096 token rows
  const size_t NQ   = (size_t)H_ * DH_;         // 2048
  const size_t NKV  = (size_t)HK_ * DH_;        // 512

  __bf16* xb   = (__bf16*)alloc(M * HID_ * 2);
  __bf16* wqb  = (__bf16*)alloc(NQ * HID_ * 2);
  __bf16* wkb  = (__bf16*)alloc(NKV * HID_ * 2);
  __bf16* wvb  = (__bf16*)alloc(NKV * HID_ * 2);
  __bf16* wob  = (__bf16*)alloc((size_t)HID_ * NQ * 2);
  __bf16* q0   = (__bf16*)alloc(M * NQ * 2);
  __bf16* k0   = (__bf16*)alloc(M * NKV * 2);
  __bf16* v0   = (__bf16*)alloc(M * NKV * 2);
  __bf16* qt   = (__bf16*)alloc(M * NQ * 2);
  __bf16* ktt  = (__bf16*)alloc(M * NKV * 2);
  __bf16* vTt  = (__bf16*)alloc(M * NKV * 2);
  __bf16* attn = (__bf16*)alloc(M * NQ * 2);

  // --- f32 -> bf16 conversions ---
  {
    size_t n;
    n = M * HID_;      f2bf_kernel<<<(unsigned)(n / 256), 256, 0, stream>>>(x,  xb,  n);
    n = NQ * HID_;     f2bf_kernel<<<(unsigned)(n / 256), 256, 0, stream>>>(Wq, wqb, n);
    n = NKV * HID_;    f2bf_kernel<<<(unsigned)(n / 256), 256, 0, stream>>>(Wk, wkb, n);
    n = NKV * HID_;    f2bf_kernel<<<(unsigned)(n / 256), 256, 0, stream>>>(Wv, wvb, n);
    n = (size_t)HID_ * NQ;
                       f2bf_kernel<<<(unsigned)(n / 256), 256, 0, stream>>>(Wo, wob, n);
  }

  // --- QKV projections ---
  dim3 blk(256);
  gemm_bf16_kernel<false><<<dim3((unsigned)(NQ / 64),  (unsigned)(M / 128)), blk, 0, stream>>>(
      xb, wqb, q0, (int)M, (int)NQ, HID_);
  gemm_bf16_kernel<false><<<dim3((unsigned)(NKV / 64), (unsigned)(M / 128)), blk, 0, stream>>>(
      xb, wkb, k0, (int)M, (int)NKV, HID_);
  gemm_bf16_kernel<false><<<dim3((unsigned)(NKV / 64), (unsigned)(M / 128)), blk, 0, stream>>>(
      xb, wvb, v0, (int)M, (int)NKV, HID_);

  // --- RoPE + layout transforms ---
  {
    size_t n = (size_t)B_ * H_ * S_ * DH_;
    rope_kernel<<<(unsigned)(n / 256), 256, 0, stream>>>(q0, qt, H_);
    n = (size_t)B_ * HK_ * S_ * DH_;
    rope_kernel<<<(unsigned)(n / 256), 256, 0, stream>>>(k0, ktt, HK_);
    vtrans_kernel<<<(unsigned)(n / 256), 256, 0, stream>>>(v0, vTt);
  }

  // --- causal flash attention ---
  attn_kernel<<<dim3(S_ / 16, B_ * H_), dim3(32), 0, stream>>>(qt, ktt, vTt, attn);

  // --- output projection (f32 result) ---
  gemm_bf16_kernel<true><<<dim3((unsigned)(HID_ / 64), (unsigned)(M / 128)), blk, 0, stream>>>(
      attn, wob, d_out, (int)M, HID_, (int)NQ);
}
